// SlidingWindowAttention_56495999812285
// MI455X (gfx1250) — matrix operand order repaired and checked by reading, not compiled, not run on hardware
//
#include <hip/hip_runtime.h>
#include <hip/hip_bf16.h>

#define DEV __device__ __forceinline__

typedef __attribute__((ext_vector_type(16))) __bf16 v16bf;
typedef __attribute__((ext_vector_type(8)))  __bf16 v8bf;
typedef __attribute__((ext_vector_type(8)))  float  v8f;

static constexpr int Bsz   = 2;
static constexpr int Lseq  = 4096;
static constexpr int Cdim  = 1024;
static constexpr int Hn    = 16;
static constexpr int Dh    = 64;      // head dim
static constexpr int WIN   = 128;
static constexpr int NB    = Lseq / WIN;       // 32 blocks
static constexpr int LP    = Lseq + 2 * WIN;   // 4352 padded length
static constexpr int MROWS = Bsz * Lseq;       // 8192
static constexpr int NTA   = 18;               // score tiles per wave (288-col window)
static constexpr int PWID  = NTA * 16;         // 288
static constexpr int LDSR  = 40;               // B-panel LDS row stride in bf16 (80 B, conflict-free)

// ---------- bf16 helpers ----------
DEV __bf16 f2bf(float f) {
    unsigned u = __builtin_bit_cast(unsigned, f);
    unsigned r = u + 0x7FFFu + ((u >> 16) & 1u);   // round-to-nearest-even
    unsigned short h = (unsigned short)(r >> 16);
    return __builtin_bit_cast(__bf16, h);
}

// ---------- CDNA5 async global->LDS copy (ASYNCcnt path), inline asm ----------
DEV void async_copy_b128(unsigned ldsOff, const void* g) {
    asm volatile("global_load_async_to_lds_b128 %0, %1, off"
                 :: "v"(ldsOff), "v"(g) : "memory");
}
DEV void wait_async0() {
    asm volatile("s_wait_asynccnt 0x0" ::: "memory");
}

// ---------- WMMA fragment loaders (CDNA5 wave32 layouts, ISA 7.12.2) ----------
// A matrix 16x32 (MxK), row-major source with leading dim lda (elements).
DEV v16bf frag_a(const __bf16* base, int lda) {
    const int lane = threadIdx.x & 31;
    const __bf16* p = base + (size_t)(lane & 15) * lda + ((lane < 16) ? 0 : 8);
    v8bf lo = *reinterpret_cast<const v8bf*>(p);
    v8bf hi = *reinterpret_cast<const v8bf*>(p + 16);
    v16bf r;
#pragma unroll
    for (int e = 0; e < 8; ++e) { r[e] = lo[e]; r[e + 8] = hi[e]; }
    return r;
}

// B matrix 32x16 (KxN) supplied transposed row-major: elem(k,n)=base[n*ldb+k].
DEV v16bf frag_bt(const __bf16* base, int ldb) {
    const int lane = threadIdx.x & 31;
    const __bf16* p = base + (size_t)(lane & 15) * ldb + ((lane < 16) ? 0 : 8);
    v8bf lo = *reinterpret_cast<const v8bf*>(p);
    v8bf hi = *reinterpret_cast<const v8bf*>(p + 16);
    v16bf r;
#pragma unroll
    for (int e = 0; e < 8; ++e) { r[e] = lo[e]; r[e + 8] = hi[e]; }
    return r;
}

DEV v8f wmma_bf16(v16bf a, v16bf b, v8f c) {
    return __builtin_amdgcn_wmma_f32_16x16x32_bf16(false, a, false, b, (short)0, c,
                                                   false, false);
}

// ---------- elementwise kernels ----------
__global__ void k_cvt_bf16(const float* __restrict__ in, __bf16* __restrict__ out, int n) {
    int i = blockIdx.x * blockDim.x + threadIdx.x;
    if (i < n) out[i] = f2bf(in[i]);
}

// wt[n*K + k] = w[k*N + n]  (K == N == 1024)
__global__ void k_transpose_bf16(const float* __restrict__ w, __bf16* __restrict__ wt) {
    int i = blockIdx.x * blockDim.x + threadIdx.x;
    if (i < Cdim * Cdim) {
        int nn = i / Cdim, kk = i - nn * Cdim;
        wt[i] = f2bf(w[kk * Cdim + nn]);
    }
}

__global__ void k_zero_u32(unsigned int* __restrict__ p, int n) {
    int i = blockIdx.x * blockDim.x + threadIdx.x;
    if (i < n) p[i] = 0u;
}

// ---------- fused projection GEMM with async-LDS double-buffered B panel ----------
// 512 threads = 16 waves; 128x128 WG tile; each wave: 16 rows x 64 cols (4 acc tiles).
// Branch-free, manually 2x-unrolled software pipeline (buffers 0/1 hard-coded) so
// accumulators stay pinned in place (no WMMA->VALU copy/NOP overhead).
// mode 0: Q bf16 [B,H,L,D] | 1: K bf16 [B,H,L+2W,D] | 2: V^T bf16 [B,H,D,L+2W]
// mode 3: fp32 [M,N] (output projection)
__global__ void __launch_bounds__(512)
gemm_bf16(const __bf16* __restrict__ A, const __bf16* __restrict__ Wt,
          const float* __restrict__ bias, int mode,
          __bf16* __restrict__ obf, float* __restrict__ of32) {
    __shared__ __align__(16) __bf16 Bp[2][128 * LDSR];   // 2 x 10 KB double buffer

    const int tid = threadIdx.x;
    const int wv = tid >> 5, lane = tid & 31;
    const int m0 = blockIdx.y * 128 + (wv & 7) * 16;     // 8 row strips
    const int nh = (wv >> 3) * 64;                       // 2 column halves
    const int n0 = blockIdx.x * 128;

    v8f acc[4];
#pragma unroll
    for (int t = 0; t < 4; ++t)
#pragma unroll
        for (int j = 0; j < 8; ++j) acc[t][j] = 0.0f;

    const __bf16* arow = A + (size_t)m0 * Cdim;

    // one async 16-B copy per thread stages a full 128x32 bf16 panel of Wt
    const int sr = tid >> 2, sq = tid & 3;
    const __bf16* gstage = Wt + (size_t)(n0 + sr) * Cdim + sq * 8;
    const unsigned lds0 = (unsigned)(uintptr_t)&Bp[0][sr * LDSR + sq * 8];
    const unsigned lds1 = (unsigned)(uintptr_t)&Bp[1][sr * LDSR + sq * 8];

    auto stage = [&](int kb, unsigned ldsOff) {
        async_copy_b128(ldsOff, gstage + kb);
    };
    auto compute = [&](int kb, int buf) {
        v16bf a = frag_a(arow + kb, Cdim);
        __builtin_prefetch(arow + kb + 64, 0, 1);           // global_prefetch_b8
#pragma unroll
        for (int nt = 0; nt < 4; ++nt) {
            v16bf b = frag_bt(&Bp[buf][(nh + nt * 16) * LDSR], LDSR);  // ds_load_b128 x2
            acc[nt] = wmma_bf16(a, b, acc[nt]);
        }
    };

    stage(0, lds0);
    wait_async0();
    __syncthreads();

    int kb = 0;
#pragma unroll 1
    for (int it = 0; it < (Cdim / 64) - 1; ++it, kb += 64) {
        stage(kb + 32, lds1);
        compute(kb, 0);
        wait_async0();
        __syncthreads();
        stage(kb + 64, lds0);
        compute(kb + 32, 1);
        wait_async0();
        __syncthreads();
    }
    // tail pair (kb == Cdim - 64): no further staging after the first half
    stage(kb + 32, lds1);
    compute(kb, 0);
    wait_async0();
    __syncthreads();
    compute(kb + 32, 1);

    const int nl = lane & 15, mh = (lane < 16) ? 0 : 8;
#pragma unroll
    for (int nt = 0; nt < 4; ++nt) {
#pragma unroll
        for (int j = 0; j < 8; ++j) {
            const int m = m0 + mh + j;
            const int n = n0 + nh + nt * 16 + nl;
            const float v = acc[nt][j] + bias[n];
            const int bb = m >> 12, l = m & (Lseq - 1);
            const int h = n >> 6, d = n & 63;
            if (mode == 0)
                obf[((size_t)(bb * Hn + h) * Lseq + l) * Dh + d] = f2bf(v);
            else if (mode == 1)
                obf[((size_t)(bb * Hn + h) * LP + (l + WIN)) * Dh + d] = f2bf(v);
            else if (mode == 2)
                obf[((size_t)(bb * Hn + h) * Dh + d) * LP + (l + WIN)] = f2bf(v);
            else
                of32[(size_t)m * Cdim + n] = v;
        }
    }
}

// ---------- banded attention: one WG per (block, head, batch) ----------
// Each wave owns a 16-row strip and computes only the 288-col (18-tile, 32-aligned)
// window that covers its 272-col band. Scores in 18 f32 accumulators -> mask ->
// softmax (16-lane shfl reductions) -> P bf16 in per-wave LDS -> O = P @ V.
__global__ void __launch_bounds__(256)
attn_kernel(const __bf16* __restrict__ q, const __bf16* __restrict__ kp,
            const __bf16* __restrict__ vt, __bf16* __restrict__ attn) {
    extern __shared__ __align__(16) __bf16 Pls[];   // 8 waves * 16 * 288 bf16 = 72 KB
    const int blk = blockIdx.x, h = blockIdx.y, bb = blockIdx.z;
    const int wv = threadIdx.x >> 5, lane = threadIdx.x & 31;
    const int m0 = wv * 16;
    const int t0 = wv & ~1;                    // first 16-col tile of this wave's window
    const int c0 = t0 * 16;                    // window column base (32-aligned)

    const __bf16* qbase = q  + ((size_t)(bb * Hn + h) * Lseq + blk * WIN + m0) * Dh;
    const __bf16* kwin  = kp + ((size_t)(bb * Hn + h) * LP + blk * WIN) * Dh;
    const __bf16* vwin  = vt + ((size_t)(bb * Hn + h) * Dh) * LP + blk * WIN + c0;

    // ---- phase 1: S = Q * K^T over this wave's 288-col window ----
    v8f s[NTA];
#pragma unroll
    for (int t = 0; t < NTA; ++t)
#pragma unroll
        for (int j = 0; j < 8; ++j) s[t][j] = 0.0f;

#pragma unroll
    for (int kb = 0; kb < Dh; kb += 32) {
        v16bf a = frag_a(qbase + kb, Dh);
#pragma unroll
        for (int nt = 0; nt < NTA; ++nt) {
            v16bf b = frag_bt(kwin + (size_t)((t0 + nt) * 16) * Dh + kb, Dh);
            s[nt] = wmma_bf16(a, b, s[nt]);
        }
    }

    // ---- mask + scale + row max (rows live in one lane-half; 16-lane reduce) ----
    const int nl = lane & 15, mh = (lane < 16) ? 0 : 8;
    const float scale = 0.125f;   // 1/sqrt(64)
    float mx[8];
#pragma unroll
    for (int j = 0; j < 8; ++j) mx[j] = -3.0e38f;
#pragma unroll
    for (int nt = 0; nt < NTA; ++nt) {
#pragma unroll
        for (int j = 0; j < 8; ++j) {
            const int col = c0 + nt * 16 + nl;
            const int m   = m0 + mh + j;
            const bool ok = (col >= m) && (col <= m + 2 * WIN);
            const float v = ok ? s[nt][j] * scale : -1.0e30f;
            s[nt][j] = v;
            mx[j] = fmaxf(mx[j], v);
        }
    }
#pragma unroll
    for (int j = 0; j < 8; ++j) {
#pragma unroll
        for (int off = 1; off < 16; off <<= 1)
            mx[j] = fmaxf(mx[j], __shfl_xor(mx[j], off, 32));
    }

    // ---- exp, row sum, stage P (bf16) into per-wave LDS ----
    __bf16* Pw = Pls + (size_t)wv * 16 * PWID;
    float sm[8];
#pragma unroll
    for (int j = 0; j < 8; ++j) sm[j] = 0.0f;
#pragma unroll
    for (int nt = 0; nt < NTA; ++nt) {
#pragma unroll
        for (int j = 0; j < 8; ++j) {
            const float e = __expf(s[nt][j] - mx[j]);
            sm[j] += e;
            Pw[(mh + j) * PWID + nt * 16 + nl] = f2bf(e);
        }
    }
    float rinv[8];
#pragma unroll
    for (int j = 0; j < 8; ++j) {
#pragma unroll
        for (int off = 1; off < 16; off <<= 1)
            sm[j] += __shfl_xor(sm[j], off, 32);
        rinv[j] = 1.0f / sm[j];
    }

    // ---- phase 2: O = P @ V over same window (Vt: elem(k,d)=vwin[d*LP+k]) ----
    v8f o[4];
#pragma unroll
    for (int t = 0; t < 4; ++t)
#pragma unroll
        for (int j = 0; j < 8; ++j) o[t][j] = 0.0f;

#pragma unroll 3
    for (int kb = 0; kb < PWID; kb += 32) {
        v16bf a = frag_a(Pw + kb, PWID);
#pragma unroll
        for (int nt = 0; nt < 4; ++nt) {
            v16bf b = frag_bt(vwin + (size_t)(nt * 16) * LP + kb, LP);
            o[nt] = wmma_bf16(a, b, o[nt]);
        }
    }

    // ---- scaled store into attn buffer [B, L, H*D] (bf16, feeds output GEMM) ----
#pragma unroll
    for (int nt = 0; nt < 4; ++nt) {
#pragma unroll
        for (int j = 0; j < 8; ++j) {
            const int m = m0 + mh + j;
            const int d = nt * 16 + nl;
            const float v = o[nt][j] * rinv[j];
            attn[((size_t)(bb * Lseq + blk * WIN + m)) * Cdim + h * Dh + d] = f2bf(v);
        }
    }
}

// ---------- host orchestration ----------
extern "C" void kernel_launch(void* const* d_in, const int* in_sizes, int n_in,
                              void* d_out, int out_size, void* d_ws, size_t ws_size,
                              hipStream_t stream) {
    const float* x  = (const float*)d_in[0];
    const float* wq = (const float*)d_in[1];
    const float* bq = (const float*)d_in[2];
    const float* wk = (const float*)d_in[3];
    const float* bk = (const float*)d_in[4];
    const float* wv = (const float*)d_in[5];
    const float* bv = (const float*)d_in[6];
    const float* wo = (const float*)d_in[7];
    const float* bo = (const float*)d_in[8];

    char* ws = (char*)d_ws;
    size_t off = 0;
    auto alloc = [&](size_t bytes) -> void* {
        void* p = ws + off;
        off += (bytes + 255) & ~(size_t)255;
        return p;
    };
    __bf16* xb   = (__bf16*)alloc((size_t)MROWS * Cdim * 2);
    __bf16* wqt  = (__bf16*)alloc((size_t)Cdim * Cdim * 2);
    __bf16* wkt  = (__bf16*)alloc((size_t)Cdim * Cdim * 2);
    __bf16* wvt  = (__bf16*)alloc((size_t)Cdim * Cdim * 2);
    __bf16* wot  = (__bf16*)alloc((size_t)Cdim * Cdim * 2);
    __bf16* qh   = (__bf16*)alloc((size_t)Bsz * Hn * Lseq * Dh * 2);
    __bf16* kpad = (__bf16*)alloc((size_t)Bsz * Hn * LP * Dh * 2);
    __bf16* vtp  = (__bf16*)alloc((size_t)Bsz * Hn * Dh * LP * 2);
    __bf16* attn = (__bf16*)alloc((size_t)MROWS * Cdim * 2);
    (void)ws_size; (void)in_sizes; (void)n_in; (void)out_size;

    const int nx = MROWS * Cdim;
    k_cvt_bf16<<<(nx + 255) / 256, 256, 0, stream>>>(x, xb, nx);

    const int nw = Cdim * Cdim;
    k_transpose_bf16<<<(nw + 255) / 256, 256, 0, stream>>>(wq, wqt);
    k_transpose_bf16<<<(nw + 255) / 256, 256, 0, stream>>>(wk, wkt);
    k_transpose_bf16<<<(nw + 255) / 256, 256, 0, stream>>>(wv, wvt);
    k_transpose_bf16<<<(nw + 255) / 256, 256, 0, stream>>>(wo, wot);

    const int nz = (int)((size_t)Bsz * Hn * LP * Dh * 2 / 4);   // u32 count
    k_zero_u32<<<(nz + 255) / 256, 256, 0, stream>>>((unsigned int*)kpad, nz);
    k_zero_u32<<<(nz + 255) / 256, 256, 0, stream>>>((unsigned int*)vtp, nz);

    dim3 gg(Cdim / 128, MROWS / 128);   // (8, 64)
    gemm_bf16<<<gg, 512, 0, stream>>>(xb, wqt, bq, 0, qh, nullptr);
    gemm_bf16<<<gg, 512, 0, stream>>>(xb, wkt, bk, 1, kpad, nullptr);
    gemm_bf16<<<gg, 512, 0, stream>>>(xb, wvt, bv, 2, vtp, nullptr);

    attn_kernel<<<dim3(NB, Hn, Bsz), 256, 8 * 16 * PWID * 2, stream>>>(qh, kpad, vtp, attn);

    gemm_bf16<<<gg, 512, 0, stream>>>(attn, wot, bo, 3, nullptr, (float*)d_out);
}
